// VectorQuantizer_17617955848573
// MI455X (gfx1250) — matrix operand order repaired and checked by reading, not compiled, not run on hardware
//
#include <hip/hip_runtime.h>
#include <hip/hip_bf16.h>

typedef __bf16 v16bf __attribute__((ext_vector_type(16)));
typedef __bf16 v8bf  __attribute__((ext_vector_type(8)));
typedef float  v8f   __attribute__((ext_vector_type(8)));

union BF16x16 { v16bf v; v8bf h[2]; };

#define VQ_B   32
#define VQ_D   256
#define VQ_HW  1024
#define VQ_N   32768      // B*HW
#define VQ_K   8192
#define VQ_NUMEL 8388608  // B*D*HW

// ---------------------------------------------------------------------------
// 1) z [B, D, HW] (f32)  ->  Zbf [N, D] (bf16), N index = b*HW + hw
// ---------------------------------------------------------------------------
__global__ void vq_transpose_z(const float* __restrict__ z, __bf16* __restrict__ Zbf) {
  __shared__ float tile[32][33];
  const int b   = blockIdx.z;
  const int hw0 = blockIdx.x * 32;
  const int d0  = blockIdx.y * 32;
  const int tx  = threadIdx.x, ty = threadIdx.y;
  const float* zb = z + (size_t)b * VQ_D * VQ_HW;
#pragma unroll
  for (int j = 0; j < 32; j += 8)
    tile[ty + j][tx] = zb[(size_t)(d0 + ty + j) * VQ_HW + hw0 + tx];
  __syncthreads();
  __bf16* dst = Zbf + (size_t)b * VQ_HW * VQ_D;
#pragma unroll
  for (int j = 0; j < 32; j += 8)
    dst[(size_t)(hw0 + ty + j) * VQ_D + d0 + tx] = (__bf16)tile[tx][ty + j];
}

// ---------------------------------------------------------------------------
// 2) embedding [D, K] (f32) -> Ebf [K, D] (bf16) and Et [K, D] (f32)
// ---------------------------------------------------------------------------
__global__ void vq_transpose_e(const float* __restrict__ emb,
                               __bf16* __restrict__ Ebf, float* __restrict__ Et) {
  __shared__ float tile[32][33];
  const int k0 = blockIdx.x * 32;
  const int d0 = blockIdx.y * 32;
  const int tx = threadIdx.x, ty = threadIdx.y;
#pragma unroll
  for (int j = 0; j < 32; j += 8)
    tile[ty + j][tx] = emb[(size_t)(d0 + ty + j) * VQ_K + k0 + tx];
  __syncthreads();
#pragma unroll
  for (int j = 0; j < 32; j += 8) {
    float v = tile[tx][ty + j];
    size_t o = (size_t)(k0 + ty + j) * VQ_D + d0 + tx;
    Et[o]  = v;
    Ebf[o] = (__bf16)v;
  }
}

// ---------------------------------------------------------------------------
// 3) enorm[k] = sum_d emb[d][k]^2 ; also zero the loss accumulator
// ---------------------------------------------------------------------------
__global__ void vq_enorm(const float* __restrict__ emb, float* __restrict__ enorm,
                         float* __restrict__ lossacc) {
  const int k = blockIdx.x * blockDim.x + threadIdx.x;
  if (k == 0) *lossacc = 0.0f;
  float s = 0.0f;
#pragma unroll 4
  for (int d = 0; d < VQ_D; ++d) {
    float v = emb[(size_t)d * VQ_K + k];
    s += v * v;
  }
  enorm[k] = s;
}

// ---------------------------------------------------------------------------
// 4) Fused GEMM + argmin, double-buffered.
//    256 blocks x 256 threads (8 waves). Wave w owns rows n0 = blk*128 + w*16
//    with A resident in VGPRs. Per 16-code chunk: issue next chunk's global
//    loads to registers, compute current chunk from LDS buf[c&1] with
//    software-pipelined ds_load_b128 B fragments + 8x v_wmma_f32_16x16x32_bf16,
//    then store the staged registers to buf[c&1 ^ 1]. ONE barrier per chunk.
//    LDS entry stride padded to 528 B -> conflict-free b128 reads.
// ---------------------------------------------------------------------------
__global__ void __launch_bounds__(256) vq_argmin(
    const __bf16* __restrict__ Zbf, const __bf16* __restrict__ Ebf,
    const float* __restrict__ enorm, int* __restrict__ kmin) {
  __shared__ __align__(16) char ldsE[2][16 * 528];  // 2 x 8448 B

  const int tid  = threadIdx.x;
  const int lane = tid & 31;
  const int wave = tid >> 5;          // 0..7
  const int half = lane >> 4;         // 0 or 1
  const int r    = lane & 15;
  const int n0   = blockIdx.x * 128 + wave * 16;

  // Staging assignment: this thread copies bytes [o0, o0+16) and [o1, o1+16)
  // of each 8192-byte chunk (16 codes x 512 B).
  const int o0 = tid * 16;
  const int o1 = o0 + 4096;
  const int l0 = (o0 >> 9) * 528 + (o0 & 511);   // padded LDS offset
  const int l1 = (o1 >> 9) * 528 + (o1 & 511);

  // A fragments: 16 rows x 256 depth, resident for the whole K loop.
  // Lane half 0: K = s*32 + {0..15}; half 1: {16..31}.
  BF16x16 afrag[8];
  {
    const __bf16* arow = Zbf + (size_t)(n0 + r) * VQ_D;
#pragma unroll
    for (int s = 0; s < 8; ++s) {
      const v8bf* p = (const v8bf*)(arow + s * 32 + half * 16);
      afrag[s].h[0] = p[0];   // +0
      afrag[s].h[1] = p[1];   // +8
    }
  }

  // Prologue: stage chunk 0 into buffer 0.
  {
    const char* src = (const char*)Ebf;
    *(v8bf*)(ldsE[0] + l0) = *(const v8bf*)(src + o0);
    *(v8bf*)(ldsE[0] + l1) = *(const v8bf*)(src + o1);
  }
  __syncthreads();

  float minv[8];
  int   mini[8];
#pragma unroll
  for (int i = 0; i < 8; ++i) { minv[i] = 3.4e38f; mini[i] = 0; }

  for (int c = 0; c < VQ_K / 16; ++c) {
    const int kb = c * 16;
    char* buf  = ldsE[c & 1];
    char* nbuf = ldsE[(c & 1) ^ 1];

    // Issue next chunk's global loads early (consumed at bottom of iteration).
    const bool more = (c + 1) < (VQ_K / 16);
    v8bf r0, r1;
    if (more) {
      const char* src = (const char*)(Ebf + (size_t)(kb + 16) * VQ_D);
      r0 = *(const v8bf*)(src + o0);
      r1 = *(const v8bf*)(src + o1);
      __builtin_prefetch(src + 8192 + o0, 0, 0);  // warm L2 two chunks ahead
    }
    const float en = enorm[kb + r];
    const int   kk = kb + r;

    // Compute: software-pipelined B fragments (lane = code r; half picks
    // depth 0-15 / 16-31 of each 32-deep step).
    v8f cc = {};
    BF16x16 bb[2];
    {
      const v8bf* bp = (const v8bf*)(buf + r * 528 + half * 32);
      bb[0].h[0] = bp[0];
      bb[0].h[1] = bp[1];
    }
#pragma unroll
    for (int s = 0; s < 8; ++s) {
      if (s < 7) {
        const v8bf* bp = (const v8bf*)(buf + r * 528 + (s + 1) * 64 + half * 32);
        bb[(s + 1) & 1].h[0] = bp[0];
        bb[(s + 1) & 1].h[1] = bp[1];
      }
      cc = __builtin_amdgcn_wmma_f32_16x16x32_bf16(
               false, afrag[s].v, false, bb[s & 1].v, (short)0, cc, false, false);
    }

    // dist = ||e||^2 - 2 z.e   (||z||^2 constant per row -> dropped)
#pragma unroll
    for (int i = 0; i < 8; ++i) {
      const float dist = __builtin_fmaf(-2.0f, cc[i], en);
      const bool lt = dist < minv[i];
      minv[i] = lt ? dist : minv[i];
      mini[i] = lt ? kk   : mini[i];
    }

    // Stage next chunk into the other buffer; single barrier per iteration:
    // reads of buf[x] (iter c) precede this barrier; writes to buf[x] occur
    // in iter c+1 strictly after it.
    if (more) {
      *(v8bf*)(nbuf + l0) = r0;
      *(v8bf*)(nbuf + l1) = r1;
    }
    __syncthreads();
  }

  // Cross-lane argmin within each 16-lane group (C layout: lane = column,
  // VGPR slot i = row i (lanes 0-15) or row 8+i (lanes 16-31)).
#pragma unroll
  for (int m = 8; m >= 1; m >>= 1) {
#pragma unroll
    for (int i = 0; i < 8; ++i) {
      const float od = __shfl_xor(minv[i], m, 32);
      const int   oi = __shfl_xor(mini[i], m, 32);
      if (od < minv[i] || (od == minv[i] && oi < mini[i])) {
        minv[i] = od; mini[i] = oi;
      }
    }
  }

  if (r == 0) {
#pragma unroll
    for (int i = 0; i < 8; ++i)
      kmin[n0 + half * 8 + i] = mini[i];
  }
}

// ---------------------------------------------------------------------------
// 5) Gather f32 codebook rows, emit [B, D, HW] output, accumulate SSE.
// ---------------------------------------------------------------------------
__global__ void __launch_bounds__(256) vq_gather(
    const float* __restrict__ z, const float* __restrict__ Et,
    const int* __restrict__ kmin, float* __restrict__ out,
    float* __restrict__ lossacc) {
  __shared__ float wsum[8];
  const int n  = blockIdx.x;
  const int d  = threadIdx.x;
  const int b  = n >> 10;
  const int hw = n & 1023;
  const int k  = kmin[n];

  const float q  = Et[(size_t)k * VQ_D + d];
  const size_t zi = (size_t)b * (VQ_D * VQ_HW) + (size_t)d * VQ_HW + hw;
  const float zv = z[zi];
  out[zi] = q;                       // quantized_st == quantized numerically

  const float diff = q - zv;
  float v = diff * diff;
#pragma unroll
  for (int m = 16; m >= 1; m >>= 1) v += __shfl_xor(v, m, 32);
  if ((threadIdx.x & 31) == 0) wsum[threadIdx.x >> 5] = v;
  __syncthreads();
  if (threadIdx.x == 0) {
    float s = 0.0f;
#pragma unroll
    for (int w = 0; w < 8; ++w) s += wsum[w];
    atomicAdd(lossacc, s);
  }
}

// ---------------------------------------------------------------------------
// 6) loss = (1 + 0.25) * SSE / numel  (e_latent == q_latent numerically)
// ---------------------------------------------------------------------------
__global__ void vq_finalize(const float* __restrict__ lossacc, float* __restrict__ out) {
  out[VQ_NUMEL] = 1.25f * (*lossacc) / (float)VQ_NUMEL;
}

extern "C" void kernel_launch(void* const* d_in, const int* in_sizes, int n_in,
                              void* d_out, int out_size, void* d_ws, size_t ws_size,
                              hipStream_t stream) {
  const float* z   = (const float*)d_in[0];   // [32, 256, 32, 32]
  const float* emb = (const float*)d_in[1];   // [256, 8192]
  float* out = (float*)d_out;                 // [8388608 quantized] + [1 loss]
  char*  ws  = (char*)d_ws;

  // Workspace layout (bytes)
  __bf16* Zbf   = (__bf16*)(ws);                    // 32768*256*2 = 16 MB
  __bf16* Ebf   = (__bf16*)(ws + 16777216);         // 8192*256*2  =  4 MB
  float*  Et    = (float*) (ws + 20971520);         // 8192*256*4  =  8 MB
  float*  enorm = (float*) (ws + 29360128);         // 8192*4      = 32 KB
  int*    kmin  = (int*)   (ws + 29392896);         // 32768*4     = 128 KB
  float*  loss  = (float*) (ws + 29523968);         // 4 B

  vq_transpose_z<<<dim3(VQ_HW / 32, VQ_D / 32, VQ_B), dim3(32, 8), 0, stream>>>(z, Zbf);
  vq_transpose_e<<<dim3(VQ_K / 32, VQ_D / 32), dim3(32, 8), 0, stream>>>(emb, Ebf, Et);
  vq_enorm<<<VQ_K / 256, 256, 0, stream>>>(emb, enorm, loss);
  vq_argmin<<<VQ_N / 128, 256, 0, stream>>>(Zbf, Ebf, enorm, kmin);
  vq_gather<<<VQ_N, VQ_D, 0, stream>>>(z, Et, kmin, out, loss);
  vq_finalize<<<1, 1, 0, stream>>>(loss, out);
}
